// GatedSlotAttention_54503134986380
// MI455X (gfx1250) — matrix operand order repaired and checked by reading, not compiled, not run on hardware
//
#include <hip/hip_runtime.h>
#include <hip/hip_bf16.h>

// ---------------------------------------------------------------------------
// Gated Slot Attention for MI455X (gfx1250).
//  Phase 0: one-time fp32 -> bf16 conversion of x and all weight matrices
//  Phase 1: bf16 WMMA GEMM, double-buffered async-LDS pipeline, fused epilogue
//  Phase 2: sequential gated-slot scan, one workgroup per (b,h), state in VGPRs
//  Phase 3: output projection (same WMMA GEMM, identity activation)
// ---------------------------------------------------------------------------

typedef __attribute__((ext_vector_type(16))) __bf16 bf16x16;
typedef __attribute__((ext_vector_type(8)))  float  v8f;
typedef __attribute__((ext_vector_type(4)))  int    v4i;

struct alignas(16) U4 { unsigned int x, y, z, w; };
union FragU { U4 u[2]; bf16x16 v; };

#if __has_builtin(__builtin_amdgcn_global_load_async_to_lds_b128) && \
    __has_builtin(__builtin_amdgcn_s_wait_asynccnt)
#define HAS_ASYNC_LDS 1
#else
#define HAS_ASYNC_LDS 0
#endif

static __device__ inline unsigned short f2bf(float f) {
    unsigned int u = __float_as_uint(f);
    unsigned int r = u + 0x7FFFu + ((u >> 16) & 1u);   // round-to-nearest-even
    return (unsigned short)(r >> 16);
}
static __device__ inline unsigned int pack2(float a, float b) {
    return (unsigned int)f2bf(a) | ((unsigned int)f2bf(b) << 16);
}

#if HAS_ASYNC_LDS
static __device__ inline void async_cp16(const void* g, void* l) {
    __builtin_amdgcn_global_load_async_to_lds_b128(
        (__attribute__((address_space(1))) v4i*)(v4i*)(void*)g,
        (__attribute__((address_space(3))) v4i*)(v4i*)l,
        0, 0);
}
#endif

#define GDIM 1024
#define BM   128
#define BN   128
#define KC   64
#define NKIT (GDIM / KC)   // 16 K-iterations
#define LDSU 36            // uints per LDS row: 32 data (64 bf16) + 4 pad

// ---------------------------------------------------------------------------
// fp32 -> packed bf16 conversion (vectorized, grid-stride)
// ---------------------------------------------------------------------------
__global__ void __launch_bounds__(256)
f32_to_bf16(const float* __restrict__ src, unsigned short* __restrict__ dst, int n4)
{
    int i = blockIdx.x * blockDim.x + threadIdx.x;
    const int stride = gridDim.x * blockDim.x;
    const float4* s4 = (const float4*)src;
    uint2* d2 = (uint2*)dst;
    for (; i < n4; i += stride) {
        float4 f = s4[i];
        d2[i] = make_uint2(pack2(f.x, f.y), pack2(f.z, f.w));
    }
}

// ---------------------------------------------------------------------------
// One K-tile of WMMAs (two 16x16x32 steps), fragments per ISA 7.12.2 layouts.
// ---------------------------------------------------------------------------
static __device__ inline void wmma_tile(const unsigned int* __restrict__ Asb,
                                        const unsigned int* __restrict__ Bsb,
                                        int mW, int nW, int half, int lidx,
                                        v8f (&acc)[2][4])
{
#pragma unroll
    for (int kk = 0; kk < 2; ++kk) {
        const int koff = kk * 16;   // uints
        bf16x16 afrag[2];
#pragma unroll
        for (int mt = 0; mt < 2; ++mt) {
            const unsigned int* base = Asb + (mW + mt * 16 + lidx) * LDSU + koff;
            FragU f;
            f.u[0] = *(const U4*)(base + half * 4);
            f.u[1] = *(const U4*)(base + 8 + half * 4);
            afrag[mt] = f.v;
        }
        bf16x16 bfrag[4];
#pragma unroll
        for (int nt = 0; nt < 4; ++nt) {
            const unsigned int* base = Bsb + (nW + nt * 16 + lidx) * LDSU + koff + half * 4;
            FragU f;
            f.u[0] = *(const U4*)(base);
            f.u[1] = *(const U4*)(base + 8);
            bfrag[nt] = f.v;
        }
#pragma unroll
        for (int mt = 0; mt < 2; ++mt)
#pragma unroll
            for (int nt = 0; nt < 4; ++nt)
                acc[mt][nt] = __builtin_amdgcn_wmma_f32_16x16x32_bf16(
                    false, afrag[mt], false, bfrag[nt],
                    (short)0, acc[mt][nt], false, false);
    }
}

// ---------------------------------------------------------------------------
// Y[n, o] = act( sum_d A[n,d] * B[o,d] + bias[o] ), A/B bf16, accum fp32
// act: 0 = silu, 1 = sigmoid^10, 2 = identity
// ---------------------------------------------------------------------------
__global__ void __launch_bounds__(256)
gemm_bf16(const unsigned short* __restrict__ A, const unsigned short* __restrict__ B,
          const float* __restrict__ bias, float* __restrict__ Y, int act)
{
    // double-buffered tiles: 2 x (18 + 18) KB = 72 KB (WGP has 320 KB LDS)
    __shared__ __align__(16) unsigned int As[2][BM * LDSU];
    __shared__ __align__(16) unsigned int Bs[2][BN * LDSU];

    const int tid  = threadIdx.x;
    const int lane = tid & 31;
    const int wave = tid >> 5;          // 8 waves
    const int mW   = (wave >> 1) * 32;  // wave tile: 32 rows x 64 cols
    const int nW   = (wave & 1) * 64;
    const int rowB = blockIdx.x * BM;
    const int colB = blockIdx.y * BN;

    const int ldRow = tid >> 1;   // 0..127
    const int ldSeg = tid & 1;    // each thread stages 32 contiguous bf16 (64 B)

    const int half = lane >> 4;
    const int lidx = lane & 15;

    const unsigned short* aBase = A + (size_t)(rowB + ldRow) * GDIM + ldSeg * 32;
    const unsigned short* bBase = B + (size_t)(colB + ldRow) * GDIM + ldSeg * 32;
    const int ldsOff = ldRow * LDSU + ldSeg * 16;

    v8f acc[2][4];
#pragma unroll
    for (int mt = 0; mt < 2; ++mt)
#pragma unroll
        for (int nt = 0; nt < 4; ++nt)
#pragma unroll
            for (int e = 0; e < 8; ++e) acc[mt][nt][e] = 0.0f;

#if HAS_ASYNC_LDS
    // prologue: fill buffer 0
#pragma unroll
    for (int j = 0; j < 4; ++j) {
        async_cp16(aBase + j * 8, As[0] + ldsOff + j * 4);
        async_cp16(bBase + j * 8, Bs[0] + ldsOff + j * 4);
    }
#pragma unroll 2
    for (int i = 0; i < NKIT; ++i) {
        const int c = i & 1;
        __builtin_amdgcn_s_wait_asynccnt(0);   // our copies into buffer c done
        __syncthreads();                       // everyone's copies done / readers done
        if (i + 1 < NKIT) {                    // overlap: fill buffer c^1
            const int kn = (i + 1) * KC;
#pragma unroll
            for (int j = 0; j < 4; ++j) {
                async_cp16(aBase + kn + j * 8, As[c ^ 1] + ldsOff + j * 4);
                async_cp16(bBase + kn + j * 8, Bs[c ^ 1] + ldsOff + j * 4);
            }
            if (i + 2 < NKIT) {                // warm L2 for the slab after that
                __builtin_prefetch(aBase + (i + 2) * KC, 0, 3);
                __builtin_prefetch(bBase + (i + 2) * KC, 0, 3);
            }
        }
        wmma_tile(As[c], Bs[c], mW, nW, half, lidx, acc);
    }
#else
    for (int i = 0; i < NKIT; ++i) {
        const int k0 = i * KC;
        U4 av[4], bv[4];
#pragma unroll
        for (int j = 0; j < 4; ++j) {
            av[j] = ((const U4*)(aBase + k0))[j];
            bv[j] = ((const U4*)(bBase + k0))[j];
        }
        __syncthreads();
#pragma unroll
        for (int j = 0; j < 4; ++j) {
            ((U4*)(As[0] + ldsOff))[j] = av[j];
            ((U4*)(Bs[0] + ldsOff))[j] = bv[j];
        }
        __syncthreads();
        wmma_tile(As[0], Bs[0], mW, nW, half, lidx, acc);
    }
#endif

    // Epilogue. D layout: VGPR r -> row = r + 8*half, col = lidx (ISA 7.12.2)
#pragma unroll
    for (int mt = 0; mt < 2; ++mt)
#pragma unroll
        for (int nt = 0; nt < 4; ++nt) {
            const int col = colB + nW + nt * 16 + lidx;
            const float bc = bias[col];
#pragma unroll
            for (int r = 0; r < 8; ++r) {
                const int row = rowB + mW + mt * 16 + half * 8 + r;
                float vv = acc[mt][nt][r] + bc;
                if (act == 0) {                                       // silu
                    vv = vv * __builtin_amdgcn_rcpf(1.0f + __expf(-vv));
                } else if (act == 1) {                                // sigmoid^10
                    float s  = __builtin_amdgcn_rcpf(1.0f + __expf(-vv));
                    float s2 = s * s, s4 = s2 * s2, s8 = s4 * s4;
                    vv = s8 * s2;
                }
                Y[(size_t)row * GDIM + col] = vv;
            }
        }
}

// ---------------------------------------------------------------------------
// Sequential scan: one workgroup (256 threads) per (b,h), 2 barriers per step.
// Thread (m = tid>>2, quarter = tid&3) owns Ke[m][16q..16q+15], Ve[...] in VGPRs.
// Softmax is computed redundantly per-wave via shuffles (no serialization).
// Output written directly as bf16 (only consumed by the final bf16 GEMM).
// ---------------------------------------------------------------------------
#define SLEN 2048
#define TT   16    // timesteps staged per LDS tile

__global__ void __launch_bounds__(256)
gsa_scan(const float* __restrict__ qp, const float* __restrict__ kp,
         const float* __restrict__ vp, const float* __restrict__ ap,
         unsigned short* __restrict__ op)
{
    const int bh = blockIdx.x;
    const int b  = bh >> 4;
    const int h  = bh & 15;
    const int tid     = threadIdx.x;
    const int lane    = tid & 31;
    const int wave    = tid >> 5;
    const int m       = tid >> 2;   // slot row 0..63
    const int quarter = tid & 3;    // d-strip 0..3

    __shared__ __align__(16) float4 sq[TT][16];
    __shared__ __align__(16) float4 sk[TT][16];
    __shared__ __align__(16) float4 sv[TT][16];
    __shared__ __align__(16) float  sa[TT][64];
    __shared__ __align__(16) float  ob[TT][64];
    __shared__ __align__(16) float  o_part[8][64];
    __shared__ float e_buf[64];

    float ke[16], ve[16];
#pragma unroll
    for (int d = 0; d < 16; ++d) { ke[d] = 0.0f; ve[d] = 0.0f; }

    const int tLd  = tid >> 4;   // staging: step index 0..15
    const int c4Ld = tid & 15;   // staging: float4 column 0..15
    const int srcl = m & 31;     // shuffle source lane for p[m]
    const bool loHalf = (m < 32);

    for (int s0 = 0; s0 < SLEN; s0 += TT) {
        __syncthreads();
        {   // stage TT steps of q/k/v/alpha into LDS (one float4 per thread per array)
            const size_t tok = ((size_t)(b * SLEN + s0 + tLd)) * GDIM + h * 64;
            sq[tLd][c4Ld] = *((const float4*)(qp + tok) + c4Ld);
            sk[tLd][c4Ld] = *((const float4*)(kp + tok) + c4Ld);
            sv[tLd][c4Ld] = *((const float4*)(vp + tok) + c4Ld);
            ((float4*)&sa[tLd][0])[c4Ld] = *((const float4*)(ap + tok) + c4Ld);
        }
        __syncthreads();

        for (int tt = 0; tt < TT; ++tt) {
            // --- state update + energy partial ---
            const float a  = sa[tt][m];
            const float om = 1.0f - a;
            float e = 0.0f;
#pragma unroll
            for (int j = 0; j < 4; ++j) {
                const float4 kk = sk[tt][quarter * 4 + j];
                const float4 qq = sq[tt][quarter * 4 + j];
                const float4 vv = sv[tt][quarter * 4 + j];
                const int d = 4 * j;
                ke[d + 0] = a * ke[d + 0] + om * kk.x; e = fmaf(ke[d + 0], qq.x, e);
                ke[d + 1] = a * ke[d + 1] + om * kk.y; e = fmaf(ke[d + 1], qq.y, e);
                ke[d + 2] = a * ke[d + 2] + om * kk.z; e = fmaf(ke[d + 2], qq.z, e);
                ke[d + 3] = a * ke[d + 3] + om * kk.w; e = fmaf(ke[d + 3], qq.w, e);
                ve[d + 0] = a * ve[d + 0] + om * vv.x;
                ve[d + 1] = a * ve[d + 1] + om * vv.y;
                ve[d + 2] = a * ve[d + 2] + om * vv.z;
                ve[d + 3] = a * ve[d + 3] + om * vv.w;
            }
            // reduce energy across the 4 lanes of row m (consecutive lanes)
            e += __shfl_xor(e, 1, 32);
            e += __shfl_xor(e, 2, 32);
            if (quarter == 0) e_buf[m] = e;
            __syncthreads();                                   // B1

            // --- softmax over 64 slots, redundantly in every wave ---
            const float v0 = e_buf[lane], v1 = e_buf[lane + 32];  // LDS broadcast
            float mx = fmaxf(v0, v1);
#pragma unroll
            for (int off = 16; off > 0; off >>= 1)
                mx = fmaxf(mx, __shfl_xor(mx, off, 32));
            const float p0 = __expf(v0 - mx), p1 = __expf(v1 - mx);
            float ss = p0 + p1;
#pragma unroll
            for (int off = 16; off > 0; off >>= 1)
                ss += __shfl_xor(ss, off, 32);
            const float invs = __builtin_amdgcn_rcpf(ss);
            const float psel = loHalf ? p0 : p1;               // wave-uniform select
            const float pm   = __shfl(psel, srcl, 32) * invs;  // p[m], normalized

            // --- readout: o[d] = sum_m pm*Ve[m][d] ---
            float opart[16];
#pragma unroll
            for (int d = 0; d < 16; ++d) opart[d] = pm * ve[d];
#pragma unroll
            for (int mask = 4; mask <= 16; mask <<= 1)
#pragma unroll
                for (int d = 0; d < 16; ++d)
                    opart[d] += __shfl_xor(opart[d], mask, 32);
            if (lane < 4) {   // lane == quarter of its reduction group
                float4* dst = (float4*)&o_part[wave][lane * 16];
                dst[0] = make_float4(opart[0],  opart[1],  opart[2],  opart[3]);
                dst[1] = make_float4(opart[4],  opart[5],  opart[6],  opart[7]);
                dst[2] = make_float4(opart[8],  opart[9],  opart[10], opart[11]);
                dst[3] = make_float4(opart[12], opart[13], opart[14], opart[15]);
            }
            __syncthreads();                                   // B2
            if (tid < 64) {
                float accv = 0.0f;
#pragma unroll
                for (int w = 0; w < 8; ++w) accv += o_part[w][tid];
                ob[tt][tid] = accv;
            }
        }
        __syncthreads();
        {   // flush staged outputs as packed bf16 (uint2 = 4 elems per thread)
            const size_t tok = ((size_t)(b * SLEN + s0 + tLd)) * GDIM + h * 64 + c4Ld * 4;
            float4 f = ((const float4*)&ob[tLd][0])[c4Ld];
            ((uint2*)op)[tok >> 2] = make_uint2(pack2(f.x, f.y), pack2(f.z, f.w));
        }
    }
}

// ---------------------------------------------------------------------------
extern "C" void kernel_launch(void* const* d_in, const int* in_sizes, int n_in,
                              void* d_out, int out_size, void* d_ws, size_t ws_size,
                              hipStream_t stream)
{
    const float* x       = (const float*)d_in[0];
    const float* W_q     = (const float*)d_in[1];
    const float* b_q     = (const float*)d_in[2];
    const float* W_k     = (const float*)d_in[3];
    const float* b_k     = (const float*)d_in[4];
    const float* W_v     = (const float*)d_in[5];
    const float* b_v     = (const float*)d_in[6];
    const float* W_alpha = (const float*)d_in[7];
    const float* b_alpha = (const float*)d_in[8];
    const float* W_o     = (const float*)d_in[9];
    const float* b_o     = (const float*)d_in[10];
    float* out = (float*)d_out;

    const size_t TOK = 4 * 2048;          // 8192 tokens
    const size_t N   = TOK * GDIM;        // elems per activation buffer
    const size_t WN  = (size_t)GDIM * GDIM;

    // workspace layout (~178 MB total)
    float* q    = (float*)d_ws;
    float* k    = q + N;
    float* v    = k + N;
    float* al   = v + N;
    unsigned short* xb   = (unsigned short*)(al + N);
    unsigned short* omid = xb + N;
    unsigned short* wq   = omid + N;
    unsigned short* wk   = wq + WN;
    unsigned short* wv   = wk + WN;
    unsigned short* wal  = wv + WN;
    unsigned short* wo   = wal + WN;

    // Phase 0: bf16 conversions
    f32_to_bf16<<<2048, 256, 0, stream>>>(x,       xb,  (int)(N / 4));
    f32_to_bf16<<<512,  256, 0, stream>>>(W_q,     wq,  (int)(WN / 4));
    f32_to_bf16<<<512,  256, 0, stream>>>(W_k,     wk,  (int)(WN / 4));
    f32_to_bf16<<<512,  256, 0, stream>>>(W_v,     wv,  (int)(WN / 4));
    f32_to_bf16<<<512,  256, 0, stream>>>(W_alpha, wal, (int)(WN / 4));
    f32_to_bf16<<<512,  256, 0, stream>>>(W_o,     wo,  (int)(WN / 4));

    dim3 ggrid(TOK / BM, GDIM / BN);      // 64 x 8
    dim3 gblk(256);

    // Phase 1: projections
    gemm_bf16<<<ggrid, gblk, 0, stream>>>(xb, wq,  b_q,     q,  0);
    gemm_bf16<<<ggrid, gblk, 0, stream>>>(xb, wk,  b_k,     k,  0);
    gemm_bf16<<<ggrid, gblk, 0, stream>>>(xb, wv,  b_v,     v,  0);
    gemm_bf16<<<ggrid, gblk, 0, stream>>>(xb, wal, b_alpha, al, 1);

    // Phase 2: gated slot scan
    gsa_scan<<<64, 256, 0, stream>>>(q, k, v, al, omid);

    // Phase 3: output projection
    gemm_bf16<<<ggrid, gblk, 0, stream>>>(omid, wo, b_o, out, 2);
}
